// dlrm_19774029431531
// MI455X (gfx1250) — compile-verified, matrix operand order checked by reading, not run on hardware
//
#include <hip/hip_runtime.h>
#include <hip/hip_bf16.h>
#include <hip/hip_fp16.h>

// ---------------------------------------------------------------------------
// DLRM forward for MI455X (gfx1250, wave32, WMMA).
//
// Roofline: embedding gather ~273MB -> ~12us @ 23.3TB/s (bandwidth floor).
// GEMM compute ~17 GFLOP in f16 WMMA (f32 accumulate).  Weights are
// pre-packed into WMMA B-fragment order (each lane's 16 halves contiguous,
// coalesced b128 loads).  Each wave computes a 16x64 output strip
// (4 accumulators, A fragment reused 4x) with a software-pipelined K loop
// (prefetch ks+1 before computing ks); the loop is unrolled 2x so the
// ping-pong buffer rotation is renamed away instead of emitted as v_movs.
// ---------------------------------------------------------------------------

typedef __attribute__((ext_vector_type(16))) _Float16 v16h;
typedef __attribute__((ext_vector_type(8)))  _Float16 v8h;
typedef __attribute__((ext_vector_type(8)))  float    v8f;

#define BATCH     16384
#define N_DENSE   13
#define N_SPARSE  26
#define NI        27          // N_SPARSE + 1
#define VOCAB     50000
#define DD        64
#define NPAIR     351         // 27*26/2
#define RW        416         // 64 + 351 padded to mult of 32
#define TSTRIDE   68          // LDS row stride (floats) to avoid bank conflicts

// WMMA fragment index helper (CDNA5 ISA 7.12.2, wave32):
//   A (16x32): lane L holds row M=L&15;  K = (e&7) + ((e>>3)<<4) + ((L>>4)<<3)
//   B (32x16): lane L holds col N=L&15;  same K indexing (symmetric)
//   C/D:       lane L col N=L&15; element e -> row M = e + ((L>>4)<<3)
__device__ __forceinline__ int frag_k(int e, int lane) {
    return (e & 7) + ((e >> 3) << 4) + ((lane >> 4) << 3);
}

// Load one A fragment (16x32 f16) for this lane: two 16B vector loads
// written directly into the v16h storage (no element shuffles).
__device__ __forceinline__ v16h load_afrag(const _Float16* an) {
    v16h af;
    *(v8h*)&af       = *(const v8h*)(an);
    *((v8h*)&af + 1) = *(const v8h*)(an + 16);
    return af;
}

// ---------------------------------------------------------------------------
// Weight pack: f32 (Ksrc x N, row-major) -> f16 WMMA B-fragment order.
// Layout: dst[(((colTile*(K/32) + ks)*32 + lane)*16 + e]
//         = W[ks*32 + frag_k(e,lane)][colTile*16 + (lane&15)]
// Rows >= Ksrc (zero-pad, e.g. tw0 415->416) write 0.
// ---------------------------------------------------------------------------
__global__ __launch_bounds__(256) void pack_w_kernel(const float* __restrict__ src,
                                                     _Float16* __restrict__ dst,
                                                     int K, int Ksrc, int N) {
    int tid = blockIdx.x * blockDim.x + threadIdx.x;   // (ct*ksteps + ks)*32 + lane
    int ksteps = K >> 5;
    int total = (N >> 4) * ksteps * 32;
    if (tid >= total) return;
    int lane = tid & 31;
    int ks   = (tid >> 5) % ksteps;
    int ct   = (tid >> 5) / ksteps;
    int col  = (ct << 4) + (lane & 15);
    _Float16* d = dst + (size_t)tid * 16;
#pragma unroll
    for (int e = 0; e < 16; ++e) {
        int k = (ks << 5) + frag_k(e, lane);
        d[e] = (k < Ksrc) ? (_Float16)src[(size_t)k * N + col] : (_Float16)0.0f;
    }
}

// ---------------------------------------------------------------------------
// Bottom MLP layer 0: (B,13) @ (13,512) + bias, relu -> f16.
// ---------------------------------------------------------------------------
__global__ __launch_bounds__(256) void bottom0_kernel(const float* __restrict__ dense,
                                                      const float* __restrict__ w,
                                                      const float* __restrict__ bias,
                                                      _Float16* __restrict__ out) {
    int gid = blockIdx.x * blockDim.x + threadIdx.x;     // b*512 + n
    if (gid >= BATCH * 512) return;
    int b = gid >> 9;
    int n = gid & 511;
    float acc = bias[n];
    const float* drow = dense + (size_t)b * N_DENSE;
#pragma unroll
    for (int k = 0; k < N_DENSE; ++k)
        acc = fmaf(drow[k], w[k * 512 + n], acc);
    out[gid] = (_Float16)fmaxf(acc, 0.0f);
}

// ---------------------------------------------------------------------------
// WMMA GEMM: Out(B x N) = relu?( A(B x K) @ W(K x N) + bias ), N%64==0,
// K%32==0.  One wave -> 16x64 output strip, software-pipelined K loop.
// ---------------------------------------------------------------------------
__global__ __launch_bounds__(256) void wmma_gemm_kernel(const _Float16* __restrict__ A,
                                                        const _Float16* __restrict__ Wp,
                                                        const float* __restrict__ bias,
                                                        _Float16* __restrict__ Out,
                                                        int K, int N, int totalWaves,
                                                        int doRelu) {
    int wave = (blockIdx.x * blockDim.x + threadIdx.x) >> 5;
    if (wave >= totalWaves) return;
    int lane = threadIdx.x & 31;
    int lmod = lane & 15;
    int lhk  = (lane >> 4) << 3;

    int strips    = N >> 6;                  // 64-col strips
    int rowBase   = (wave / strips) << 4;
    int tileBase  = (wave % strips) << 2;    // first 16-col tile of the strip
    int ksteps    = K >> 5;

    v8f c[4] = {{}, {}, {}, {}};
    const _Float16* arow = A + (size_t)(rowBase + lmod) * K + lhk;
    // Per-tile packed-W fragment pointers (include +lane*16); one K-step
    // advances 32 lanes * 16 halves = 512 halves = 32 v16h elements.
    const v16h* wp[4];
#pragma unroll
    for (int t = 0; t < 4; ++t)
        wp[t] = (const v16h*)(Wp + (((size_t)(tileBase + t) * ksteps << 5) + lane) * 16);

    // ---- prologue: load K-step 0 ----
    v16h afc = load_afrag(arow);
    v16h bfc[4];
#pragma unroll
    for (int t = 0; t < 4; ++t) bfc[t] = wp[t][0];

    // Software pipeline, unrolled 2x so buffer rotation is register-renamed.
#pragma unroll 2
    for (int ks = 0; ks < ksteps; ++ks) {
        // ---- prefetch K-step ks+1 (clamped; last iter redundantly reloads) ----
        int ksn = (ks + 1 < ksteps) ? ks + 1 : ks;
        v16h afn = load_afrag(arow + (ksn << 5));
        v16h bfn[4];
#pragma unroll
        for (int t = 0; t < 4; ++t) bfn[t] = wp[t][(size_t)ksn * 32];

        // ---- compute on current fragments (previous iteration's loads) ----
#pragma unroll
        for (int t = 0; t < 4; ++t)
            c[t] = __builtin_amdgcn_wmma_f32_16x16x32_f16(
                       false, afc, false, bfc[t], (short)0, c[t], false, false);
        afc = afn;
#pragma unroll
        for (int t = 0; t < 4; ++t) bfc[t] = bfn[t];
    }

#pragma unroll
    for (int t = 0; t < 4; ++t) {
        int col = ((tileBase + t) << 4) + lmod;
        float bv = (bias != nullptr) ? bias[col] : 0.0f;
#pragma unroll
        for (int e = 0; e < 8; ++e) {
            float v = c[t][e] + bv;
            if (doRelu) v = fmaxf(v, 0.0f);
            Out[(size_t)(rowBase + e + lhk) * N + col] = (_Float16)v;
        }
    }
}

// ---------------------------------------------------------------------------
// Interaction: one wave per batch row.  Gather T = [x ; emb rows] (27x64 f32)
// into LDS, then emit R[b] = [x (f16, 64) | tril_{k=-1}(T T^T) (351) | 0 pad].
// ---------------------------------------------------------------------------
__global__ __launch_bounds__(256) void interact_kernel(const float* __restrict__ emb,
                                                       const int* __restrict__ sidx,
                                                       const _Float16* __restrict__ x,
                                                       _Float16* __restrict__ R) {
    __shared__ float T[8][NI * TSTRIDE];
    int wave = threadIdx.x >> 5;
    int lane = threadIdx.x & 31;
    int b    = blockIdx.x * 8 + wave;

    float* Tw = T[wave];
    const _Float16* xrow = x + (size_t)b * DD;

    for (int t = lane; t < DD; t += 32)
        Tw[t] = (float)xrow[t];
    for (int f = 0; f < N_SPARSE; ++f) {
        int idx = sidx[b * N_SPARSE + f];
        const float* erow = emb + ((size_t)f * VOCAB + idx) * DD;
        for (int t = lane; t < DD; t += 32)
            Tw[(f + 1) * TSTRIDE + t] = erow[t];
    }
    __syncthreads();

    _Float16* Rrow = R + (size_t)b * RW;
    for (int t = lane; t < DD; t += 32)
        Rrow[t] = xrow[t];
    if (lane == 0)
        Rrow[RW - 1] = (_Float16)0.0f;   // pad column 415

    for (int p = lane; p < NPAIR; p += 32) {
        int i = 1;
        while ((i * (i + 1)) / 2 <= p) ++i;
        int j = p - (i * (i - 1)) / 2;
        const float* Ti = Tw + i * TSTRIDE;
        const float* Tj = Tw + j * TSTRIDE;
        float acc = 0.0f;
#pragma unroll
        for (int d = 0; d < DD; ++d)
            acc = fmaf(Ti[d], Tj[d], acc);
        Rrow[DD + p] = (_Float16)acc;
    }
}

// ---------------------------------------------------------------------------
// Final layer: logits[b] = t1[b,:] @ tw2 (256x1).
// ---------------------------------------------------------------------------
__global__ __launch_bounds__(256) void final_kernel(const _Float16* __restrict__ t1,
                                                    const float* __restrict__ tw2,
                                                    float* __restrict__ out) {
    int wave = threadIdx.x >> 5;
    int lane = threadIdx.x & 31;
    int b    = blockIdx.x * 8 + wave;
    const _Float16* row = t1 + (size_t)b * 256;
    float acc = 0.0f;
#pragma unroll
    for (int i = 0; i < 8; ++i) {
        int k = lane + i * 32;
        acc = fmaf((float)row[k], tw2[k], acc);
    }
#pragma unroll
    for (int off = 16; off > 0; off >>= 1)
        acc += __shfl_xor(acc, off, 32);
    if (lane == 0) out[b] = acc;
}

// ---------------------------------------------------------------------------
// Host launcher
// ---------------------------------------------------------------------------
extern "C" void kernel_launch(void* const* d_in, const int* in_sizes, int n_in,
                              void* d_out, int out_size, void* d_ws, size_t ws_size,
                              hipStream_t stream) {
    const float* dense = (const float*)d_in[0];
    const int*   sidx  = (const int*)  d_in[1];
    const float* emb   = (const float*)d_in[2];
    const float* bw0   = (const float*)d_in[3];
    const float* bb0   = (const float*)d_in[4];
    const float* bw1   = (const float*)d_in[5];
    const float* bb1   = (const float*)d_in[6];
    const float* bw2   = (const float*)d_in[7];
    const float* bb2   = (const float*)d_in[8];
    const float* tw0   = (const float*)d_in[9];
    const float* tw1   = (const float*)d_in[10];
    const float* tw2   = (const float*)d_in[11];
    float* out = (float*)d_out;

    char* ws = (char*)d_ws;
    size_t off = 0;
    _Float16* h0   = (_Float16*)(ws + off); off += (size_t)BATCH * 512 * 2;  // also t0
    _Float16* h1   = (_Float16*)(ws + off); off += (size_t)BATCH * 256 * 2;  // also t1
    _Float16* x    = (_Float16*)(ws + off); off += (size_t)BATCH * DD  * 2;
    _Float16* R    = (_Float16*)(ws + off); off += (size_t)BATCH * RW  * 2;
    _Float16* w1p  = (_Float16*)(ws + off); off += (size_t)512 * 256 * 2;
    _Float16* w2p  = (_Float16*)(ws + off); off += (size_t)256 * DD  * 2;
    _Float16* tw0p = (_Float16*)(ws + off); off += (size_t)RW  * 512 * 2;
    _Float16* tw1p = (_Float16*)(ws + off); off += (size_t)512 * 256 * 2;
    _Float16* t0 = h0;
    _Float16* t1 = h1;

    {   int thr = (256 / 16) * (512 / 32) * 32;
        pack_w_kernel<<<(thr + 255) / 256, 256, 0, stream>>>(bw1, w1p, 512, 512, 256); }
    {   int thr = (64 / 16) * (256 / 32) * 32;
        pack_w_kernel<<<(thr + 255) / 256, 256, 0, stream>>>(bw2, w2p, 256, 256, 64); }
    {   int thr = (512 / 16) * (RW / 32) * 32;
        pack_w_kernel<<<(thr + 255) / 256, 256, 0, stream>>>(tw0, tw0p, RW, 415, 512); }
    {   int thr = (256 / 16) * (512 / 32) * 32;
        pack_w_kernel<<<(thr + 255) / 256, 256, 0, stream>>>(tw1, tw1p, 512, 512, 256); }

    bottom0_kernel<<<(BATCH * 512) / 256, 256, 0, stream>>>(dense, bw0, bb0, h0);

    {   int waves = (BATCH / 16) * (256 / 64);
        wmma_gemm_kernel<<<waves / 8, 256, 0, stream>>>(h0, w1p, bb1, h1, 512, 256, waves, 1); }
    {   int waves = (BATCH / 16) * (DD / 64);
        wmma_gemm_kernel<<<waves / 8, 256, 0, stream>>>(h1, w2p, bb2, x, 256, DD, waves, 1); }

    interact_kernel<<<BATCH / 8, 256, 0, stream>>>(emb, sidx, x, R);

    {   int waves = (BATCH / 16) * (512 / 64);
        wmma_gemm_kernel<<<waves / 8, 256, 0, stream>>>(R, tw0p, nullptr, t0, RW, 512, waves, 1); }
    {   int waves = (BATCH / 16) * (256 / 64);
        wmma_gemm_kernel<<<waves / 8, 256, 0, stream>>>(t0, tw1p, nullptr, t1, 512, 256, waves, 1); }

    final_kernel<<<BATCH / 8, 256, 0, stream>>>(t1, tw2, out);
    (void)in_sizes; (void)n_in; (void)out_size; (void)ws_size;
}
